// Video_multilevel_encoding_with_support_41214506172610
// MI455X (gfx1250) — compile-verified, matrix operand order checked by reading, not run on hardware
//
#include <hip/hip_runtime.h>
#include <hip/hip_bf16.h>
#include <math.h>

// ---------------------------------------------------------------------------
// MI455X (gfx1250) implementation. wave32, WMMA bf16 16x16x32, f32 accum.
//   1) cast inputs/weights -> bf16 (WMMA-friendly [N][K] layouts)
//   2) xg GEMM: 128x128 block tiles, BK=64, LDS double-buffered via
//      GLOBAL_LOAD_ASYNC_TO_LDS_B128 (ASYNCcnt)
//   3) GRU scan: 48 serial steps, h in LDS, gate triples fused in registers
//   4) masked mean pool + time-padded gm + feature concat
//   5) conv1d (w=2..5): gm slab async-staged to LDS once per (seq,w); implicit
//      im2col WMMA GEMM + relu + atomic max-pool
//   6) attention projections (WMMA) + softmax gate + residual output
// ---------------------------------------------------------------------------

typedef __attribute__((ext_vector_type(16))) __bf16 v16bf;
typedef __attribute__((ext_vector_type(8)))  __bf16 v8bf;
typedef __attribute__((ext_vector_type(8)))  float  v8f;
typedef __attribute__((ext_vector_type(4)))  int    v4i;

#define TT    48
#define DD    2048
#define HH    512
#define NSEQ  320
#define FEATN 5120
#define TPAD  72
#define CSTR  1032            // conv LDS row stride (1024 + 8 pad, elements)

// ---- async global->LDS helpers --------------------------------------------
// builtin signature (probe-confirmed via diagnostic): (v4i AS1*, v4i AS3*, Imm, Imm)
#if __has_builtin(__builtin_amdgcn_global_load_async_to_lds_b128)
#define ASYNC_B128(g, l)                                                      \
  __builtin_amdgcn_global_load_async_to_lds_b128(                             \
      (__attribute__((address_space(1))) v4i*)(v4i*)(g),                      \
      (__attribute__((address_space(3))) v4i*)(v4i*)(l), 0, 0)
#define HAVE_ASYNC_LDS 1
#else
#define ASYNC_B128(g, l) (*(v8bf*)(l) = *(const v8bf*)(g))
#define HAVE_ASYNC_LDS 0
#endif

__device__ __forceinline__ void wait_async8() {
#if __has_builtin(__builtin_amdgcn_s_wait_asynccnt)
  __builtin_amdgcn_s_wait_asynccnt(8);
#elif HAVE_ASYNC_LDS
  asm volatile("s_wait_asynccnt 0x8" ::: "memory");
#endif
}
__device__ __forceinline__ void wait_async0() {
#if __has_builtin(__builtin_amdgcn_s_wait_asynccnt)
  __builtin_amdgcn_s_wait_asynccnt(0);
#elif HAVE_ASYNC_LDS
  asm volatile("s_wait_asynccnt 0x0" ::: "memory");
#endif
}

// ---- WMMA fragment helpers (layouts per CDNA5 ISA 7.12.2) ------------------
__device__ __forceinline__ v16bf frag_combine(v8bf lo, v8bf hi) {
  v16bf o;
#pragma unroll
  for (int i = 0; i < 8; ++i) { o[i] = lo[i]; o[i + 8] = hi[i]; }
  return o;
}

__device__ __forceinline__ v16bf load_frag(const __bf16* __restrict__ tile,
                                           int ldk, int k0, int lane) {
  const int r = lane & 15;
  const int koff = (lane >> 4) << 3;
  const __bf16* p = tile + (size_t)r * ldk + k0 + koff;
  return frag_combine(*(const v8bf*)p, *(const v8bf*)(p + 16));
}

__device__ __forceinline__ float sigmoidf_(float x) {
  return 1.0f / (1.0f + __expf(-x));
}

// ---- generic cast / pack kernels -------------------------------------------
__global__ void cast_bf16_kernel(const float* __restrict__ s,
                                 __bf16* __restrict__ d, size_t n) {
  size_t i = (size_t)blockIdx.x * blockDim.x + threadIdx.x;
  size_t st = (size_t)gridDim.x * blockDim.x;
  for (; i < n; i += st) d[i] = (__bf16)s[i];
}

__global__ void pack_bias_kernel(const float* __restrict__ bf,
                                 const float* __restrict__ bb,
                                 float* __restrict__ dst) {
  int i = blockIdx.x * blockDim.x + threadIdx.x;
  if (i < 1536) { dst[i] = bf[i]; dst[1536 + i] = bb[i]; }
}

// conv_w [512][1024][w] -> [512][w*1024] with k = tap*1024 + c
__global__ void pack_convw_kernel(const float* __restrict__ src,
                                  __bf16* __restrict__ dst, int w) {
  size_t n = (size_t)512 * 1024 * w;
  size_t i = (size_t)blockIdx.x * blockDim.x + threadIdx.x;
  size_t st = (size_t)gridDim.x * blockDim.x;
  for (; i < n; i += st) {
    int o   = (int)(i / (1024 * w));
    int rem = (int)(i % (1024 * w));
    int tap = rem / 1024;
    int c   = rem % 1024;
    dst[(size_t)o * (1024 * w) + tap * 1024 + c] =
        (__bf16)src[((size_t)o * 1024 + c) * w + tap];
  }
}

// ---- simple per-wave GEMM (small shapes only: query projection) ------------
__global__ void gemm_bf16_wmma(const __bf16* __restrict__ A,
                               const __bf16* __restrict__ Bt,
                               const float* __restrict__ bias,
                               float* __restrict__ D,
                               int M, int N, int K) {
  const int lane = threadIdx.x & 31;
  const int wid  = threadIdx.x >> 5;
  const int wpb  = blockDim.x >> 5;
  const int tilesN = N >> 4;
  const int total  = (M >> 4) * tilesN;
  for (int t = blockIdx.x * wpb + wid; t < total; t += gridDim.x * wpb) {
    const int tm = t / tilesN, tn = t % tilesN;
    const __bf16* Atile = A + (size_t)tm * 16 * K;
    const __bf16* Btile = Bt + (size_t)tn * 16 * K;
    v8f acc = {};
    for (int k0 = 0; k0 < K; k0 += 32) {
      __builtin_prefetch(Btile + (size_t)(lane & 15) * K + k0 + 512, 0, 1);
      v16bf a = load_frag(Atile, K, k0, lane);
      v16bf b = load_frag(Btile, K, k0, lane);
      acc = __builtin_amdgcn_wmma_f32_16x16x32_bf16(false, a, false, b,
                                                    (short)0, acc, false, false);
    }
    const int n    = (tn << 4) + (lane & 15);
    const int mrow = (tm << 4) + ((lane >> 4) << 3);
    const float bv = bias ? bias[n] : 0.0f;
#pragma unroll
    for (int r = 0; r < 8; ++r)
      D[(size_t)(mrow + r) * N + n] = acc[r] + bv;
  }
}

// ---- block-tiled GEMM: BM=BN=128, BK=64, async double-buffered LDS ---------
#define BM 128
#define BN 128
#define BK 64
#define LSTR 72                       // BK + 8 pad, elements
#define LBUF (128 * LSTR)             // elements per A/B stage buffer

// copy 128 rows x 64 cols bf16 (row stride ldg) into LDS (stride LSTR);
// 4 x b128 per thread, 8 per thread counting A+B
__device__ __forceinline__ void stage_tile(const __bf16* __restrict__ g, int ldg,
                                           __bf16* l, int tid, int k0) {
  const int seg  = tid & 7;           // 8 x 8-element segments per row
  const int row0 = tid >> 3;          // 0..31
#pragma unroll
  for (int j = 0; j < 4; ++j) {
    const int row = row0 + j * 32;
    ASYNC_B128(g + (size_t)row * ldg + k0 + seg * 8, l + row * LSTR + seg * 8);
  }
}

__global__ void gemm_bf16_wmma_lds(const __bf16* __restrict__ A,
                                   const __bf16* __restrict__ Bt,
                                   const float* __restrict__ bias,
                                   float* __restrict__ D,
                                   int M, int N, int K) {
  extern __shared__ __align__(16) __bf16 smem[];
  __bf16* Ab0 = smem;
  __bf16* Bb0 = smem + LBUF;
  __bf16* Ab1 = smem + 2 * LBUF;
  __bf16* Bb1 = smem + 3 * LBUF;

  const int tid  = threadIdx.x;
  const int lane = tid & 31;
  const int wid  = tid >> 5;
  const int wm   = wid >> 1;          // 0..3: 32-row region
  const int wn   = wid & 1;           // 0..1: 64-col region
  const int bn   = blockIdx.x % (N / BN);
  const int bm   = blockIdx.x / (N / BN);
  const __bf16* Ag = A + (size_t)bm * BM * K;
  const __bf16* Bg = Bt + (size_t)bn * BN * K;
  const int nK = K / BK;

  v8f acc[2][4];
  v8f z = {};
#pragma unroll
  for (int i = 0; i < 2; ++i)
#pragma unroll
    for (int j = 0; j < 4; ++j) acc[i][j] = z;

  stage_tile(Ag, K, Ab0, tid, 0);
  stage_tile(Bg, K, Bb0, tid, 0);

  for (int kb = 0; kb < nK; ++kb) {
    __bf16* Ac = (kb & 1) ? Ab1 : Ab0;
    __bf16* Bc = (kb & 1) ? Bb1 : Bb0;
    if (kb + 1 < nK) {
      __bf16* An = (kb & 1) ? Ab0 : Ab1;
      __bf16* Bn = (kb & 1) ? Bb0 : Bb1;
      stage_tile(Ag, K, An, tid, (kb + 1) * BK);
      stage_tile(Bg, K, Bn, tid, (kb + 1) * BK);
      wait_async8();                  // current stage's 8 ops retired
    } else {
      wait_async0();
    }
    __syncthreads();
#pragma unroll
    for (int kt = 0; kt < 2; ++kt) {
      v16bf af[2], bfr[4];
#pragma unroll
      for (int i = 0; i < 2; ++i)
        af[i] = load_frag(Ac + (wm * 32 + i * 16) * LSTR, LSTR, kt * 32, lane);
#pragma unroll
      for (int j = 0; j < 4; ++j)
        bfr[j] = load_frag(Bc + (wn * 64 + j * 16) * LSTR, LSTR, kt * 32, lane);
#pragma unroll
      for (int i = 0; i < 2; ++i)
#pragma unroll
        for (int j = 0; j < 4; ++j)
          acc[i][j] = __builtin_amdgcn_wmma_f32_16x16x32_bf16(
              false, af[i], false, bfr[j], (short)0, acc[i][j], false, false);
    }
    __syncthreads();
  }

  const int nc = lane & 15;
  const int mr = (lane >> 4) << 3;
#pragma unroll
  for (int i = 0; i < 2; ++i)
#pragma unroll
    for (int j = 0; j < 4; ++j) {
      const int gn = bn * BN + wn * 64 + j * 16 + nc;
      const int gm = bm * BM + wm * 32 + i * 16 + mr;
      const float bv = bias ? bias[gn] : 0.0f;
#pragma unroll
      for (int r = 0; r < 8; ++r)
        D[(size_t)(gm + r) * N + gn] = acc[i][j][r] + bv;
    }
}

// ---- GRU scan --------------------------------------------------------------
__global__ void gru_scan_kernel(const __bf16* __restrict__ Whh,  // [2][1536][512]
                                const float* __restrict__ bhh_f,
                                const float* __restrict__ bhh_b,
                                const float* __restrict__ xg,    // [NSEQ*48][3072]
                                float* __restrict__ gru) {       // [NSEQ][48][1024]
  __shared__ __align__(16) __bf16 hbf[16 * HH];
  __shared__ float hf[16 * HH];

  const int lane = threadIdx.x & 31;
  const int wid  = threadIdx.x >> 5;
  const int dir  = blockIdx.x / 20;
  const int s0   = (blockIdx.x % 20) << 4;
  const __bf16* W   = Whh + (size_t)dir * 1536 * HH;
  const float*  bhh = dir ? bhh_b : bhh_f;

  for (int i = threadIdx.x; i < 16 * HH; i += blockDim.x) {
    hbf[i] = (__bf16)0.0f;
    hf[i]  = 0.0f;
  }
  __syncthreads();

  const int ncol  = lane & 15;
  const int mbase = (lane >> 4) << 3;

  for (int t = 0; t < TT; ++t) {
    const int tio = dir ? (TT - 1 - t) : t;
    v8f zero = {};
    v8f accR[4] = {zero, zero, zero, zero};
    v8f accZ[4] = {zero, zero, zero, zero};
    v8f accN[4] = {zero, zero, zero, zero};

    for (int k0 = 0; k0 < HH; k0 += 32) {
      v16bf a = load_frag(hbf, HH, k0, lane);
#pragma unroll
      for (int jt = 0; jt < 4; ++jt) {
        const int tR = wid * 4 + jt;
        v16bf bR = load_frag(W + (size_t)tR * 16 * HH, HH, k0, lane);
        v16bf bZ = load_frag(W + (size_t)(tR + 32) * 16 * HH, HH, k0, lane);
        v16bf bN = load_frag(W + (size_t)(tR + 64) * 16 * HH, HH, k0, lane);
        accR[jt] = __builtin_amdgcn_wmma_f32_16x16x32_bf16(false, a, false, bR,
                                                           (short)0, accR[jt], false, false);
        accZ[jt] = __builtin_amdgcn_wmma_f32_16x16x32_bf16(false, a, false, bZ,
                                                           (short)0, accZ[jt], false, false);
        accN[jt] = __builtin_amdgcn_wmma_f32_16x16x32_bf16(false, a, false, bN,
                                                           (short)0, accN[jt], false, false);
      }
    }
    __syncthreads();

#pragma unroll
    for (int jt = 0; jt < 4; ++jt) {
      const int j = ((wid * 4 + jt) << 4) + ncol;
#pragma unroll
      for (int r = 0; r < 8; ++r) {
        const int mm  = mbase + r;
        const int seq = s0 + mm;
        const float* xp = xg + ((size_t)seq * TT + tio) * 3072 + dir * 1536;
        const float hr = accR[jt][r] + bhh[j];
        const float hz = accZ[jt][r] + bhh[512 + j];
        const float hn = accN[jt][r] + bhh[1024 + j];
        const float rg = sigmoidf_(xp[j] + hr);
        const float zg = sigmoidf_(xp[512 + j] + hz);
        const float ng = tanhf(xp[1024 + j] + rg * hn);
        const float hprev = hf[mm * HH + j];
        const float hnew  = (1.0f - zg) * ng + zg * hprev;
        hf[mm * HH + j]  = hnew;
        hbf[mm * HH + j] = (__bf16)hnew;
        gru[((size_t)seq * TT + tio) * 1024 + dir * 512 + j] = hnew;
      }
    }
    __syncthreads();
  }
}

// ---- masked mean pool + padded gm + feature concat -------------------------
__global__ void pool_pad_kernel(const float* __restrict__ gru,
                                const float* __restrict__ vmask,
                                const int* __restrict__ vlen,
                                const float* __restrict__ smask,
                                const int* __restrict__ slen,
                                const float* __restrict__ vorig,
                                const float* __restrict__ sorig,
                                float* __restrict__ feat,
                                __bf16* __restrict__ gmPad) {
  const int seq = blockIdx.x;
  const float* mask; float len; const float* orig;
  if (seq < 64) { mask = vmask + seq * TT; len = (float)vlen[seq]; orig = vorig + (size_t)seq * DD; }
  else { mask = smask + (seq - 64) * TT; len = (float)slen[seq - 64]; orig = sorig + (size_t)(seq - 64) * DD; }

  float* frow = feat + (size_t)seq * FEATN;
  for (int c = threadIdx.x; c < 2048; c += blockDim.x) {
    frow[1024 + c] = 0.0f;
    frow[3072 + c] = orig[c];
  }
  for (int c = threadIdx.x; c < 1024; c += blockDim.x) {
    float s = 0.0f;
    for (int t = 0; t < TT; ++t) {
      float v = gru[((size_t)seq * TT + t) * 1024 + c] * mask[t];
      s += v;
      gmPad[((size_t)seq * TPAD + (t + 4)) * 1024 + c] = (__bf16)v;
    }
    frow[c] = s / len;
    for (int t = 0; t < 4; ++t)
      gmPad[((size_t)seq * TPAD + t) * 1024 + c] = (__bf16)0.0f;
    for (int t = TT + 4; t < TPAD; ++t)
      gmPad[((size_t)seq * TPAD + t) * 1024 + c] = (__bf16)0.0f;
  }
}

// ---- conv: one block per seq; gm slab async-staged to LDS once -------------
__global__ void conv_max_wmma_lds(const __bf16* __restrict__ gmPad,
                                  const __bf16* __restrict__ Wc,   // [512][w*1024]
                                  const float* __restrict__ cbias,
                                  float* __restrict__ feat,
                                  int w, int coff) {
  extern __shared__ __align__(16) __bf16 gsl[];   // 72 * CSTR elements
  const int tid  = threadIdx.x;
  const int lane = tid & 31;
  const int wid  = tid >> 5;
  const int seq  = blockIdx.x;
  const int Kc   = w << 10;

  // stage 72 x 1024 bf16 slab: 9216 x b128, 36 per thread
  for (int u = tid; u < TPAD * 128; u += blockDim.x) {
    const int row = u >> 7;
    const int seg = u & 127;
    ASYNC_B128(gmPad + ((size_t)seq * TPAD + row) * 1024 + seg * 8,
               gsl + row * CSTR + seg * 8);
  }
  wait_async0();
  __syncthreads();

  // 4 ptiles x 32 ntiles = 128 output tiles; 16 per wave
  for (int t = wid; t < 128; t += blockDim.x >> 5) {
    const int ntile = t & 31;
    const int ptile = t >> 5;
    const int p     = (ptile << 4) + (lane & 15);
    const int koff  = (lane >> 4) << 3;
    const __bf16* Btile = Wc + (size_t)ntile * 16 * Kc;
    v8f acc = {};
    for (int kk = 0; kk < Kc; kk += 32) {
      const int tap = kk >> 10;
      const int c0  = (kk & 1023) + koff;
      const __bf16* ap = gsl + (p + tap + 5 - w) * CSTR + c0;
      v16bf a = frag_combine(*(const v8bf*)ap, *(const v8bf*)(ap + 16));
      v16bf b = load_frag(Btile, Kc, kk, lane);
      acc = __builtin_amdgcn_wmma_f32_16x16x32_bf16(false, a, false, b,
                                                    (short)0, acc, false, false);
    }
    const int n  = (ntile << 4) + (lane & 15);
    const float bv = cbias[n];
    const int pbase = (ptile << 4) + ((lane >> 4) << 3);
    float* dst = feat + (size_t)seq * FEATN + 1024 + coff + n;
#pragma unroll
    for (int r = 0; r < 8; ++r) {
      const int pp = pbase + r;
      if (pp < TT + w - 1) {
        float v = acc[r] + bv;
        v = v > 0.0f ? v : 0.0f;
        atomicMax((int*)dst, __float_as_int(v));
      }
    }
  }
}

// ---- attention: logits, softmax(S=4), gated residual -----------------------
__global__ void attn_out_kernel(const float* __restrict__ key,
                                const float* __restrict__ query,
                                const float* __restrict__ feat,
                                float* __restrict__ out) {
  const int b = blockIdx.x;
  __shared__ float red[256];
  __shared__ float wgt[4];
  for (int s = 0; s < 4; ++s) {
    float part = 0.0f;
    const float* kp = key + (size_t)(b * 4 + s) * 512;
    const float* qp = query + (size_t)b * 512;
    for (int j = threadIdx.x; j < 512; j += blockDim.x) part += kp[j] * qp[j];
    red[threadIdx.x] = part;
    __syncthreads();
    for (int o = blockDim.x >> 1; o > 0; o >>= 1) {
      if ((int)threadIdx.x < o) red[threadIdx.x] += red[threadIdx.x + o];
      __syncthreads();
    }
    if (threadIdx.x == 0) wgt[s] = red[0];
    __syncthreads();
  }
  if (threadIdx.x == 0) {
    float mx = wgt[0];
    for (int s = 1; s < 4; ++s) mx = fmaxf(mx, wgt[s]);
    float sum = 0.0f;
    for (int s = 0; s < 4; ++s) { wgt[s] = expf(wgt[s] - mx); sum += wgt[s]; }
    for (int s = 0; s < 4; ++s) wgt[s] /= sum;
  }
  __syncthreads();
  for (int f = threadIdx.x; f < FEATN; f += blockDim.x) {
    float acc = feat[(size_t)b * FEATN + f];
    for (int s = 0; s < 4; ++s)
      acc += wgt[s] * feat[(size_t)(64 + b * 4 + s) * FEATN + f];
    out[(size_t)b * FEATN + f] = acc;
  }
}

// ---------------------------------------------------------------------------
extern "C" void kernel_launch(void* const* d_in, const int* in_sizes, int n_in,
                              void* d_out, int out_size, void* d_ws, size_t ws_size,
                              hipStream_t stream) {
  const float* videos = (const float*)d_in[0];
  const float* vorig  = (const float*)d_in[1];
  const int*   vlen   = (const int*)d_in[2];
  const float* vmask  = (const float*)d_in[3];
  const float* svid   = (const float*)d_in[4];
  const float* sorig  = (const float*)d_in[5];
  const int*   slen   = (const int*)d_in[6];
  const float* smask  = (const float*)d_in[7];
  const float* Wihf = (const float*)d_in[10];
  const float* Whhf = (const float*)d_in[11];
  const float* bihf = (const float*)d_in[12];
  const float* bhhf = (const float*)d_in[13];
  const float* Wihb = (const float*)d_in[14];
  const float* Whhb = (const float*)d_in[15];
  const float* bihb = (const float*)d_in[16];
  const float* bhhb = (const float*)d_in[17];
  const float* cwsrc[4] = {(const float*)d_in[18], (const float*)d_in[20],
                           (const float*)d_in[22], (const float*)d_in[24]};
  const float* cb[4]    = {(const float*)d_in[19], (const float*)d_in[21],
                           (const float*)d_in[23], (const float*)d_in[25]};
  const float* Wk = (const float*)d_in[26];
  const float* bk = (const float*)d_in[27];
  const float* Wq = (const float*)d_in[28];
  const float* bq = (const float*)d_in[29];
  float* out = (float*)d_out;

  char* ws = (char*)d_ws;
  size_t off = 0;
  auto carve = [&](size_t bytes) {
    size_t o = off;
    off += (bytes + 255) & ~(size_t)255;
    return (void*)(ws + o);
  };
  __bf16* Xbf   = (__bf16*)carve((size_t)NSEQ * TT * DD * 2);
  __bf16* Wihbf = (__bf16*)carve((size_t)3072 * DD * 2);
  float*  bih   = (float*)carve(3072 * 4);
  __bf16* Whhbf = (__bf16*)carve((size_t)2 * 1536 * HH * 2);
  __bf16* Wcbf  = (__bf16*)carve((size_t)512 * 14336 * 2);
  __bf16* Wkbf  = (__bf16*)carve((size_t)512 * FEATN * 2);
  __bf16* Wqbf  = (__bf16*)carve((size_t)512 * FEATN * 2);
  float*  xg    = (float*)carve((size_t)NSEQ * TT * 3072 * 4);
  float*  gru   = (float*)carve((size_t)NSEQ * TT * 1024 * 4);
  __bf16* gmPad = (__bf16*)carve((size_t)NSEQ * TPAD * 1024 * 2);
  float*  feat  = (float*)carve((size_t)NSEQ * FEATN * 4);
  __bf16* fbf   = (__bf16*)carve((size_t)NSEQ * FEATN * 2);
  float*  keyb  = (float*)carve((size_t)256 * 512 * 4);
  float*  qryb  = (float*)carve((size_t)64 * 512 * 4);

  auto cblk = [](size_t n) {
    size_t b = (n + 255) / 256;
    return (unsigned)(b > 16384 ? 16384 : b);
  };

  // 1) casts / packs
  cast_bf16_kernel<<<cblk((size_t)64 * TT * DD), 256, 0, stream>>>(videos, Xbf, (size_t)64 * TT * DD);
  cast_bf16_kernel<<<cblk((size_t)256 * TT * DD), 256, 0, stream>>>(svid, Xbf + (size_t)64 * TT * DD, (size_t)256 * TT * DD);
  cast_bf16_kernel<<<cblk((size_t)1536 * DD), 256, 0, stream>>>(Wihf, Wihbf, (size_t)1536 * DD);
  cast_bf16_kernel<<<cblk((size_t)1536 * DD), 256, 0, stream>>>(Wihb, Wihbf + (size_t)1536 * DD, (size_t)1536 * DD);
  cast_bf16_kernel<<<cblk((size_t)1536 * HH), 256, 0, stream>>>(Whhf, Whhbf, (size_t)1536 * HH);
  cast_bf16_kernel<<<cblk((size_t)1536 * HH), 256, 0, stream>>>(Whhb, Whhbf + (size_t)1536 * HH, (size_t)1536 * HH);
  cast_bf16_kernel<<<cblk((size_t)512 * FEATN), 256, 0, stream>>>(Wk, Wkbf, (size_t)512 * FEATN);
  cast_bf16_kernel<<<cblk((size_t)512 * FEATN), 256, 0, stream>>>(Wq, Wqbf, (size_t)512 * FEATN);
  pack_bias_kernel<<<6, 256, 0, stream>>>(bihf, bihb, bih);
  size_t cwoff[4] = {0, (size_t)512 * 2048, (size_t)512 * 5120, (size_t)512 * 9216};
  for (int i = 0; i < 4; ++i)
    pack_convw_kernel<<<cblk((size_t)512 * 1024 * (i + 2)), 256, 0, stream>>>(
        cwsrc[i], Wcbf + cwoff[i], i + 2);

  // 2) input projection GEMM: [15360 x 2048] x [2048 -> 3072]
  gemm_bf16_wmma_lds<<<(NSEQ * TT / BM) * (3072 / BN), 256,
                       4 * LBUF * sizeof(__bf16), stream>>>(
      Xbf, Wihbf, bih, xg, NSEQ * TT, 3072, DD);

  // 3) GRU scan
  gru_scan_kernel<<<40, 256, 0, stream>>>(Whhbf, bhhf, bhhb, xg, gru);

  // 4) pool + pad + concat skeleton
  pool_pad_kernel<<<NSEQ, 256, 0, stream>>>(gru, vmask, vlen, smask, slen,
                                            vorig, sorig, feat, gmPad);

  // 5) convs: one block per seq, gm slab in LDS
  for (int i = 0; i < 4; ++i)
    conv_max_wmma_lds<<<NSEQ, 256, TPAD * CSTR * sizeof(__bf16), stream>>>(
        gmPad, Wcbf + cwoff[i], cb[i], feat, i + 2, i * 512);

  // 6) attention
  cast_bf16_kernel<<<cblk((size_t)NSEQ * FEATN), 256, 0, stream>>>(feat, fbf, (size_t)NSEQ * FEATN);
  gemm_bf16_wmma_lds<<<(256 / BM) * (512 / BN), 256,
                       4 * LBUF * sizeof(__bf16), stream>>>(
      fbf + (size_t)64 * FEATN, Wkbf, bk, keyb, 256, 512, FEATN);
  gemm_bf16_wmma<<<16, 256, 0, stream>>>(fbf, Wqbf, bq, qryb, 64, 512, FEATN);
  attn_out_kernel<<<64, 256, 0, stream>>>(keyb, qryb, feat, out);
}
